// TransformerAgent_79800492359684
// MI455X (gfx1250) — matrix operand order repaired and checked by reading, not compiled, not run on hardware
//
#include <hip/hip_runtime.h>
#include <math.h>

// ---------------- problem constants ----------------
#define Hh   64
#define Bm   16384
#define Aa   16
#define Nn   (Bm * Aa)        // 262144 rows
#define Pp   24
#define Tt   32
#define ACTD 12
#define EPSI 1e-4f

// ---------------- WMMA types / helpers ----------------
typedef __attribute__((ext_vector_type(16))) __bf16 v16bf;
typedef __attribute__((ext_vector_type(8)))  float  v8f;

// Native f32 -> bf16 conversion (v_cvt_bf16_f32 on gfx1250); avoids the
// 3-integer-op manual rounding sequence.
__device__ __forceinline__ __bf16 f2bf(float f) { return (__bf16)f; }

__device__ __forceinline__ v8f zero8() {
  v8f z;
#pragma unroll
  for (int i = 0; i < 8; ++i) z[i] = 0.0f;
  return z;
}

#define WMMA_BF16(a_, b_, c_) \
  __builtin_amdgcn_wmma_f32_16x16x32_bf16(false, (a_), false, (b_), (short)0, (c_), false, false)

// log((d+1)/(d+eps)) without the IEEE divide sequence: two v_log_f32 TRANS ops.
__device__ __forceinline__ float logratio(float d) {
  return __logf(d + 1.0f) - __logf(d + EPSI);
}

__device__ __forceinline__ float sigm(float x) { return 1.0f / (1.0f + __expf(-x)); }

// A fragment: 16 rows (M=lane&15) x K-chunk of 32, from row-major f32 (leading dim ld).
// Per ISA 16-bit A layout: lane L holds row L%16, half hl=L>>4; element e -> K = e + (e&8) + 8*hl.
__device__ __forceinline__ v16bf a_frag_rows(const float* base, int ld, int kbase) {
  int lane = threadIdx.x & 31;
  const float* r = base + (lane & 15) * ld + kbase + ((lane >> 4) << 3);
  v16bf A;
#pragma unroll
  for (int e = 0; e < 16; ++e) A[e] = f2bf(r[e + (e & 8)]);
  return A;
}

// A fragment where all 16 rows are the same 64-float vector (broadcast prototype).
__device__ __forceinline__ v16bf a_frag_bcast(const float* v, int kbase) {
  int lane = threadIdx.x & 31;
  const float* r = v + kbase + ((lane >> 4) << 3);
  v16bf A;
#pragma unroll
  for (int e = 0; e < 16; ++e) A[e] = f2bf(r[e + (e & 8)]);
  return A;
}

// B fragment: weights W[nout][K] row-major; out cols n0..n0+15, K-chunk of 32 at kbase.
// Lane L: N = n0 + L%16, element e -> K = kbase + e + (e&8) + 8*(L>>4).
__device__ __forceinline__ v16bf b_frag(const float* W, int ld, int n0, int kbase, int nmax) {
  int lane = threadIdx.x & 31;
  int n = n0 + (lane & 15);
  v16bf Bf;
  if (n < nmax) {
    const float* r = W + (size_t)n * ld + kbase + ((lane >> 4) << 3);
#pragma unroll
    for (int e = 0; e < 16; ++e) Bf[e] = f2bf(r[e + (e & 8)]);
  } else {
#pragma unroll
    for (int e = 0; e < 16; ++e) Bf[e] = f2bf(0.0f);
  }
  return Bf;
}

// ======================================================================
// K1: biLSTM over [24,32,3] -> fcl -> proj -> prototype vectors (tiny, 1 block)
// Also zeroes the BN stats accumulator.
// ======================================================================
__global__ __launch_bounds__(256) void k_proto(
    const float* __restrict__ jq,
    const float* __restrict__ w_ih_f, const float* __restrict__ w_hh_f,
    const float* __restrict__ b_ih_f, const float* __restrict__ b_hh_f,
    const float* __restrict__ w_ih_b, const float* __restrict__ w_hh_b,
    const float* __restrict__ b_ih_b, const float* __restrict__ b_hh_b,
    const float* __restrict__ fcl_w, const float* __restrict__ fcl_b,
    const float* __restrict__ proj_w,
    float* __restrict__ pvw, float* __restrict__ pvn, float* __restrict__ stats)
{
  __shared__ float hS[Pp * Hh];
  __shared__ float cS[Pp * Hh];
  __shared__ float gS[Pp * 256];
  __shared__ float thS[Pp * 128];
  __shared__ float foS[Pp * Hh];

  const int tid = threadIdx.x;
  if (tid < 128) stats[tid] = 0.0f;
  for (int i = tid; i < Pp * Hh; i += 256) { hS[i] = 0.0f; cS[i] = 0.0f; }
  __syncthreads();

  for (int t = 0; t < Tt; ++t) {
    for (int g = tid; g < Pp * 256; g += 256) {
      int p = g >> 8, q = g & 255;
      const float* xr = jq + p * (Tt * 3) + t * 3;
      float v = b_ih_f[q] + b_hh_f[q]
              + xr[0] * w_ih_f[q * 3 + 0] + xr[1] * w_ih_f[q * 3 + 1] + xr[2] * w_ih_f[q * 3 + 2];
      const float* hr = &hS[p * Hh];
      const float* wr = &w_hh_f[q * Hh];
      for (int j = 0; j < Hh; ++j) v += hr[j] * wr[j];
      gS[g] = v;
    }
    __syncthreads();
    for (int u = tid; u < Pp * Hh; u += 256) {
      int p = u >> 6, j = u & 63;
      float gi = gS[p * 256 + j], gf = gS[p * 256 + 64 + j];
      float gg = gS[p * 256 + 128 + j], go = gS[p * 256 + 192 + j];
      float c = sigm(gf) * cS[u] + sigm(gi) * tanhf(gg);
      cS[u] = c;
      hS[u] = sigm(go) * tanhf(c);
    }
    __syncthreads();
  }

  // backward direction: one step on x[:, T-1], zero initial state
  for (int g = tid; g < Pp * 256; g += 256) {
    int p = g >> 8, q = g & 255;
    const float* xr = jq + p * (Tt * 3) + (Tt - 1) * 3;
    gS[g] = b_ih_b[q] + b_hh_b[q]
          + xr[0] * w_ih_b[q * 3 + 0] + xr[1] * w_ih_b[q * 3 + 1] + xr[2] * w_ih_b[q * 3 + 2];
  }
  __syncthreads();
  for (int u = tid; u < Pp * Hh; u += 256) {
    int p = u >> 6, j = u & 63;
    thS[p * 128 + j] = hS[u];
    float gi = gS[p * 256 + j];
    float gg = gS[p * 256 + 128 + j], go = gS[p * 256 + 192 + j];
    float cb = sigm(gi) * tanhf(gg);      // c0 == 0
    thS[p * 128 + 64 + j] = sigm(go) * tanhf(cb);
  }
  __syncthreads();

  // fcl: [24,128] -> [24,64], relu
  for (int u = tid; u < Pp * Hh; u += 256) {
    int p = u >> 6, uo = u & 63;
    float v = fcl_b[uo];
    const float* tr = &thS[p * 128];
    const float* wr = &fcl_w[uo * 128];
    for (int k = 0; k < 128; ++k) v += tr[k] * wr[k];
    foS[u] = fmaxf(v, 0.0f);
  }
  __syncthreads();

  // proj (no bias): pv = fo @ proj_w.T
  for (int u = tid; u < Pp * Hh; u += 256) {
    int p = u >> 6, uo = u & 63;
    float v = 0.0f;
    const float* fr = &foS[p * Hh];
    const float* wr = &proj_w[uo * Hh];
    for (int k = 0; k < Hh; ++k) v += fr[k] * wr[k];
    pvw[u] = v;
    gS[u] = v;
  }
  __syncthreads();
  if (tid < Pp) {
    float s = 0.0f;
    for (int k = 0; k < Hh; ++k) { float v = gS[tid * Hh + k]; s += v * v; }
    pvn[tid] = s;
  }
}

// ======================================================================
// K2: per month b: subgraph distances over 6 seasonal protos x 16 agents,
// argmax of mean log-ratio -> global prototype index gidx[b].
// ======================================================================
__global__ __launch_bounds__(96) void k_choose(
    const float* __restrict__ inputs, const int* __restrict__ month,
    const float* __restrict__ pvw, int* __restrict__ gidx)
{
  __shared__ float part[96];
  __shared__ float simS[6];
  const int b = blockIdx.x, tid = threadIdx.x;
  const int season = ((month[b] + 10) / 3) & 3;
  const int j = tid >> 4, a = tid & 15;
  const float* x = inputs + ((size_t)b * Aa + a) * Hh;
  const float* p = pvw + (size_t)(season * 6 + j) * Hh;
  float d = 0.0f;
  for (int k = 0; k < Hh; ++k) { float t = x[k] - p[k]; d += t * t; }
  part[tid] = logratio(d);
  __syncthreads();
  if (tid < 6) {
    float s = 0.0f;
    for (int a2 = 0; a2 < 16; ++a2) s += part[tid * 16 + a2];
    simS[tid] = s * (1.0f / 16.0f);
  }
  __syncthreads();
  if (tid == 0) {
    int best = 0; float bs = simS[0];
    for (int jj = 1; jj < 6; ++jj) if (simS[jj] > bs) { bs = simS[jj]; best = jj; }
    gidx[b] = season * 6 + best;
  }
}

// ======================================================================
// K3a: per 16-row tile (== one month): WMMA xp -> min_distances/logits,
// WMMA conv -> per-channel sum/sumsq for BatchNorm (LDS reduce + global atomics).
// Block = 64 threads = 2 waves = 2 tiles.
// ======================================================================
__global__ __launch_bounds__(64) void k_stats(
    const float* __restrict__ inputs, const float* __restrict__ pvw,
    const float* __restrict__ pvn, const int* __restrict__ gidx,
    const float* __restrict__ conv_w, const float* __restrict__ conv_b,
    const float* __restrict__ last_w,
    float* __restrict__ out_logits, float* __restrict__ out_mind,
    float* __restrict__ stats)
{
  __shared__ float xnorm_s[2][16];
  __shared__ float act_s[2][16][24];
  __shared__ float statl[128];

  const int tid = threadIdx.x, wave = tid >> 5, lane = tid & 31;
  const int l15 = lane & 15, hl = lane >> 4;
  const long ti = (long)blockIdx.x * 2 + wave;
  const long r0 = ti * 16;
  const float* Xb = inputs + r0 * Hh;
  __builtin_prefetch(Xb, 0, 1);

  if (tid < 128) statl[tid] = 0.0f;

  v16bf ax0 = a_frag_rows(Xb, Hh, 0);
  v16bf ax1 = a_frag_rows(Xb, Hh, 32);

  if (lane < 16) {
    float s = 0.0f;
    const float* r = Xb + l15 * Hh;
    for (int k = 0; k < Hh; ++k) s += r[k] * r[k];
    xnorm_s[wave][l15] = s;
  }
  __syncthreads();

  // xp GEMM: [16,64] x [64,24] -> min_distances + activations
#pragma unroll
  for (int ct = 0; ct < 2; ++ct) {
    v8f acc = zero8();
    acc = WMMA_BF16(ax0, b_frag(pvw, Hh, ct * 16, 0, Pp), acc);
    acc = WMMA_BF16(ax1, b_frag(pvw, Hh, ct * 16, 32, Pp), acc);
    int n = ct * 16 + l15;
    if (n < Pp) {
      float pn = pvn[n];
#pragma unroll
      for (int r = 0; r < 8; ++r) {
        int mo = r + hl * 8;
        float md = -2.0f * acc[r] + xnorm_s[wave][mo] + pn;
        out_mind[(r0 + mo) * Pp + n] = md;
        act_s[wave][mo][n] = logratio(md);
      }
    }
  }
  __syncthreads();

  // logits = act @ last_w.T  (4 outputs, via LDS)
  {
    int mo = lane >> 1, ob = (lane & 1) * 2;
#pragma unroll
    for (int oo = 0; oo < 2; ++oo) {
      int o = ob + oo;
      float s = 0.0f;
      for (int jj = 0; jj < Pp; ++jj) s += act_s[wave][mo][jj] * last_w[o * Pp + jj];
      out_logits[(r0 + mo) * 4 + o] = s;
    }
  }

  // conv GEMM (for BN statistics only; result recomputed in k_main)
  const float* cv = pvw + (size_t)gidx[ti] * Hh;
  v16bf ac0 = a_frag_bcast(cv, 0);
  v16bf ac1 = a_frag_bcast(cv, 32);
#pragma unroll
  for (int ct = 0; ct < 4; ++ct) {
    v8f acc = zero8();
    acc = WMMA_BF16(ax0, b_frag(conv_w, 128, ct * 16, 0,  Hh), acc);
    acc = WMMA_BF16(ax1, b_frag(conv_w, 128, ct * 16, 32, Hh), acc);
    acc = WMMA_BF16(ac0, b_frag(conv_w, 128, ct * 16, 64, Hh), acc);
    acc = WMMA_BF16(ac1, b_frag(conv_w, 128, ct * 16, 96, Hh), acc);
    int n = ct * 16 + l15;
    float cb = conv_b[n];
    float s1 = 0.0f, s2 = 0.0f;
#pragma unroll
    for (int r = 0; r < 8; ++r) { float v = acc[r] + cb; s1 += v; s2 += v * v; }
    atomicAdd(&statl[n], s1);
    atomicAdd(&statl[64 + n], s2);
  }
  __syncthreads();
  if (tid < 128) atomicAdd(&stats[tid], statl[tid]);
}

// ======================================================================
// K3b: finalize BN (training-mode, biased var) -> folded scale/shift
// ======================================================================
__global__ __launch_bounds__(64) void k_bn(
    const float* __restrict__ stats, const float* __restrict__ bn_g,
    const float* __restrict__ bn_b, float* __restrict__ scale, float* __restrict__ shift)
{
  int c = threadIdx.x;
  const float invN = 1.0f / (float)Nn;
  float mu = stats[c] * invN;
  float var = stats[64 + c] * invN - mu * mu;
  float rs = rsqrtf(var + 1e-5f);
  float sc = bn_g[c] * rs;
  scale[c] = sc;
  shift[c] = bn_b[c] - mu * sc;
}

// ======================================================================
// K3c: fused main pass per 16-row tile:
// conv -> BN+relu -> fc1 -> LN+relu -> GRUCell -> out.  All GEMMs via bf16 WMMA.
// ======================================================================
__global__ __launch_bounds__(64) void k_main(
    const float* __restrict__ inputs, const float* __restrict__ hidden,
    const float* __restrict__ pvw, const int* __restrict__ gidx,
    const float* __restrict__ conv_w, const float* __restrict__ conv_b,
    const float* __restrict__ bnscale, const float* __restrict__ bnshift,
    const float* __restrict__ fc1_w, const float* __restrict__ fc1_b,
    const float* __restrict__ ln_g, const float* __restrict__ ln_b,
    const float* __restrict__ gru_wih, const float* __restrict__ gru_whh,
    const float* __restrict__ gru_bih, const float* __restrict__ gru_bhh,
    const float* __restrict__ out_w, const float* __restrict__ out_b,
    float* __restrict__ out_a, float* __restrict__ out_h)
{
  __shared__ float xg [2][16][64];
  __shared__ float ghl[2][16][192];
  __shared__ float rz [2][16][128];
  __shared__ float hn [2][16][64];

  const int tid = threadIdx.x, wave = tid >> 5, lane = tid & 31;
  const int l15 = lane & 15, hl = lane >> 4;
  const long ti = (long)blockIdx.x * 2 + wave;
  const long r0 = ti * 16;
  const float* Xb = inputs + r0 * Hh;
  const float* Hb = hidden + r0 * Hh;
  const float* cv = pvw + (size_t)gidx[ti] * Hh;
  __builtin_prefetch(Hb, 0, 1);

  v16bf ax0 = a_frag_rows(Xb, Hh, 0), ax1 = a_frag_rows(Xb, Hh, 32);
  v16bf ac0 = a_frag_bcast(cv, 0),    ac1 = a_frag_bcast(cv, 32);

  // conv -> BN -> relu -> LDS
#pragma unroll
  for (int ct = 0; ct < 4; ++ct) {
    v8f acc = zero8();
    acc = WMMA_BF16(ax0, b_frag(conv_w, 128, ct * 16, 0,  Hh), acc);
    acc = WMMA_BF16(ax1, b_frag(conv_w, 128, ct * 16, 32, Hh), acc);
    acc = WMMA_BF16(ac0, b_frag(conv_w, 128, ct * 16, 64, Hh), acc);
    acc = WMMA_BF16(ac1, b_frag(conv_w, 128, ct * 16, 96, Hh), acc);
    int n = ct * 16 + l15;
    float cb = conv_b[n], sc = bnscale[n], sh = bnshift[n];
#pragma unroll
    for (int r = 0; r < 8; ++r) {
      int mo = r + hl * 8;
      xg[wave][mo][n] = fmaxf((acc[r] + cb) * sc + sh, 0.0f);
    }
  }
  __syncthreads();

  // fc1 (A-frags built from LDS first, then overwrite xg with fc1 output)
  v16bf af0 = a_frag_rows(&xg[wave][0][0], Hh, 0);
  v16bf af1 = a_frag_rows(&xg[wave][0][0], Hh, 32);
  __syncthreads();
#pragma unroll
  for (int ct = 0; ct < 4; ++ct) {
    v8f acc = zero8();
    acc = WMMA_BF16(af0, b_frag(fc1_w, Hh, ct * 16, 0,  Hh), acc);
    acc = WMMA_BF16(af1, b_frag(fc1_w, Hh, ct * 16, 32, Hh), acc);
    int n = ct * 16 + l15;
    float fb = fc1_b[n];
#pragma unroll
    for (int r = 0; r < 8; ++r) xg[wave][r + hl * 8][n] = acc[r] + fb;
  }
  __syncthreads();

  // LayerNorm + relu in place (each lane: row l15, half hl of the columns)
  {
    float* row = &xg[wave][l15][0];
    float mu = 0.0f;
    for (int c = 0; c < Hh; ++c) mu += row[c];
    mu *= (1.0f / Hh);
    float var = 0.0f;
    for (int c = 0; c < Hh; ++c) { float d = row[c] - mu; var += d * d; }
    var *= (1.0f / Hh);
    float rs = rsqrtf(var + 1e-5f);
    for (int c = hl * 32; c < hl * 32 + 32; ++c)
      row[c] = fmaxf((row[c] - mu) * rs * ln_g[c] + ln_b[c], 0.0f);
  }
  __syncthreads();

  v16bf axl0 = a_frag_rows(&xg[wave][0][0], Hh, 0);
  v16bf axl1 = a_frag_rows(&xg[wave][0][0], Hh, 32);
  v16bf ah0  = a_frag_rows(Hb, Hh, 0);
  v16bf ah1  = a_frag_rows(Hb, Hh, 32);

  // gh = h_in @ gru_whh.T + b_hh  -> LDS
#pragma unroll
  for (int t = 0; t < 12; ++t) {
    v8f acc = zero8();
    acc = WMMA_BF16(ah0, b_frag(gru_whh, Hh, t * 16, 0,  192), acc);
    acc = WMMA_BF16(ah1, b_frag(gru_whh, Hh, t * 16, 32, 192), acc);
    int n = t * 16 + l15;
    float bh = gru_bhh[n];
#pragma unroll
    for (int r = 0; r < 8; ++r) ghl[wave][r + hl * 8][n] = acc[r] + bh;
  }
  __syncthreads();

  // gi tiles + GRU gate math
  for (int t = 0; t < 12; ++t) {
    v8f acc = zero8();
    acc = WMMA_BF16(axl0, b_frag(gru_wih, Hh, t * 16, 0,  192), acc);
    acc = WMMA_BF16(axl1, b_frag(gru_wih, Hh, t * 16, 32, 192), acc);
    int n = t * 16 + l15;
    float bi = gru_bih[n];
    if (t < 8) {            // r-gate (t<4) and z-gate (4<=t<8)
#pragma unroll
      for (int r = 0; r < 8; ++r) {
        int mo = r + hl * 8;
        rz[wave][mo][n] = sigm(acc[r] + bi + ghl[wave][mo][n]);
      }
    } else {                // n-gate + blend
      int cn = n - 128;
#pragma unroll
      for (int r = 0; r < 8; ++r) {
        int mo = r + hl * 8;
        float nn = tanhf(acc[r] + bi + rz[wave][mo][cn] * ghl[wave][mo][n]);
        float z  = rz[wave][mo][64 + cn];
        float hp = Hb[mo * Hh + cn];
        float hv = (1.0f - z) * nn + z * hp;
        hn[wave][mo][cn] = hv;
        out_h[(r0 + mo) * Hh + cn] = hv;
      }
    }
    __syncthreads();
  }

  // out = h_new @ out_w.T + out_b
  v16bf an0 = a_frag_rows(&hn[wave][0][0], Hh, 0);
  v16bf an1 = a_frag_rows(&hn[wave][0][0], Hh, 32);
  v8f acc = zero8();
  acc = WMMA_BF16(an0, b_frag(out_w, Hh, 0, 0,  ACTD), acc);
  acc = WMMA_BF16(an1, b_frag(out_w, Hh, 0, 32, ACTD), acc);
  if (l15 < ACTD) {
    float ob = out_b[l15];
#pragma unroll
    for (int r = 0; r < 8; ++r)
      out_a[(r0 + r + hl * 8) * ACTD + l15] = acc[r] + ob;
  }
}

// ======================================================================
extern "C" void kernel_launch(void* const* d_in, const int* in_sizes, int n_in,
                              void* d_out, int out_size, void* d_ws, size_t ws_size,
                              hipStream_t stream) {
  (void)in_sizes; (void)n_in; (void)out_size; (void)ws_size;
  const float* inputs  = (const float*)d_in[0];
  const float* hidden  = (const float*)d_in[1];
  const int*   month   = (const int*)d_in[2];
  const float* jq      = (const float*)d_in[3];
  const float* w_ih_f  = (const float*)d_in[4];
  const float* w_hh_f  = (const float*)d_in[5];
  const float* b_ih_f  = (const float*)d_in[6];
  const float* b_hh_f  = (const float*)d_in[7];
  const float* w_ih_b  = (const float*)d_in[8];
  const float* w_hh_b  = (const float*)d_in[9];
  const float* b_ih_b  = (const float*)d_in[10];
  const float* b_hh_b  = (const float*)d_in[11];
  const float* fcl_w   = (const float*)d_in[12];
  const float* fcl_b   = (const float*)d_in[13];
  const float* proj_w  = (const float*)d_in[14];
  const float* conv_w  = (const float*)d_in[15];
  const float* conv_b  = (const float*)d_in[16];
  const float* bn_g    = (const float*)d_in[17];
  const float* bn_b    = (const float*)d_in[18];
  const float* fc1_w   = (const float*)d_in[19];
  const float* fc1_b   = (const float*)d_in[20];
  const float* ln_g    = (const float*)d_in[21];
  const float* ln_b    = (const float*)d_in[22];
  const float* gru_wih = (const float*)d_in[23];
  const float* gru_whh = (const float*)d_in[24];
  const float* gru_bih = (const float*)d_in[25];
  const float* gru_bhh = (const float*)d_in[26];
  const float* out_w   = (const float*)d_in[27];
  const float* out_b   = (const float*)d_in[28];
  const float* last_w  = (const float*)d_in[29];

  // workspace layout (floats)
  float* ws      = (float*)d_ws;
  float* pvw     = ws;            // 24*64
  float* pvn     = ws + 1536;     // 24
  float* stats   = ws + 1600;     // 128 (sum, sumsq)
  float* bnscale = ws + 1728;     // 64
  float* bnshift = ws + 1792;     // 64
  int*   gidx    = (int*)(ws + 2048); // 16384 ints

  // output layout: a [N,12] | h [N,64] | logits [N,4] | min_distances [N,24]
  float* out        = (float*)d_out;
  float* out_a      = out;
  float* out_h      = out + (size_t)Nn * 12;
  float* out_logits = out + (size_t)Nn * 76;
  float* out_mind   = out + (size_t)Nn * 80;

  k_proto<<<1, 256, 0, stream>>>(jq, w_ih_f, w_hh_f, b_ih_f, b_hh_f,
                                 w_ih_b, w_hh_b, b_ih_b, b_hh_b,
                                 fcl_w, fcl_b, proj_w, pvw, pvn, stats);
  k_choose<<<Bm, 96, 0, stream>>>(inputs, month, pvw, gidx);
  k_stats<<<Nn / 32, 64, 0, stream>>>(inputs, pvw, pvn, gidx, conv_w, conv_b,
                                      last_w, out_logits, out_mind, stats);
  k_bn<<<1, 64, 0, stream>>>(stats, bn_g, bn_b, bnscale, bnshift);
  k_main<<<Nn / 32, 64, 0, stream>>>(inputs, hidden, pvw, gidx, conv_w, conv_b,
                                     bnscale, bnshift, fc1_w, fc1_b, ln_g, ln_b,
                                     gru_wih, gru_whh, gru_bih, gru_bhh,
                                     out_w, out_b, out_a, out_h);
}
